// MultiHeadedAttention_42863773614181
// MI455X (gfx1250) — matrix operand order repaired and checked by reading, not compiled, not run on hardware
//
#include <hip/hip_runtime.h>

// ---------------------------------------------------------------------------
// MI455X (gfx1250) multi-head attention, f16 WMMA path, wave32.
//
// Roofline: ~137 GFLOP vs ~150 MB HBM -> compute bound; all GEMM stages run
// on v_wmma_f32_16x16x32_f16. Weights pre-converted to f16 once (L2-resident).
// Softmax denominator is computed BY the matrix engine (P x ones-column) so
// the only cross-lane reduction left is the running max. The query-row mask
// is fused into the per-row score scale (masked row -> all-zero scores ->
// uniform softmax, identical to the reference's all -1e9 row).
// Fragment layouts used by this code (contiguous k order, not AMD's table order):
//   A (16x32 f16): lane l, g=l>>4 -> row l&15, K = g*16 + 0..15 (contiguous 16)
//   B (32x16 f16): lane l -> col n=l&15, K = (l>>4)*16 + 0..15 (contiguous 16)
//   C/D (16x16 f32): lane l, reg r -> row r+8*(l>>4), col l&15
// ---------------------------------------------------------------------------

typedef _Float16 h4  __attribute__((ext_vector_type(4)));
typedef _Float16 h8  __attribute__((ext_vector_type(8)));
typedef _Float16 h16 __attribute__((ext_vector_type(16)));
typedef float    v8f __attribute__((ext_vector_type(8)));
typedef float    f4  __attribute__((ext_vector_type(4)));

#define HID   1024
#define NH    16
#define DHD   64
#define BATCH 4
#define SEQ   2048
#define MTOT  (BATCH * SEQ)   // 8192 rows

__device__ __forceinline__ h8 cat4(h4 a, h4 b) {
  return __builtin_shufflevector(a, b, 0, 1, 2, 3, 4, 5, 6, 7);
}
__device__ __forceinline__ h16 cat8(h8 a, h8 b) {
  return __builtin_shufflevector(a, b, 0, 1, 2, 3, 4, 5, 6, 7, 8, 9, 10, 11,
                                 12, 13, 14, 15);
}
__device__ __forceinline__ v8f wmma16(h16 a, h16 b, v8f c) {
  return __builtin_amdgcn_wmma_f32_16x16x32_f16(false, a, false, b, (short)0,
                                                c, false, false);
}
__device__ __forceinline__ h16 load_b_f16(const _Float16* p) {
  return cat8(*(const h8*)p, *(const h8*)(p + 8));
}
__device__ __forceinline__ h16 splat_one16() {
  h16 v;
#pragma unroll
  for (int i = 0; i < 16; ++i) v[i] = (_Float16)1.0f;
  return v;
}
// CDNA5 async global->LDS (ASYNCcnt-tracked DMA), 16B per lane.
__device__ __forceinline__ void async_copy16(unsigned lds_off, const void* g) {
  const unsigned long long ga = (unsigned long long)g;
  asm volatile("global_load_async_to_lds_b128 %0, %1, off"
               :: "v"(lds_off), "v"(ga) : "memory");
}

// ---------------------------------------------------------------------------
// Weight pre-conversion: f32 [1024x1024] -> f16, one shot, L2-resident after.
// ---------------------------------------------------------------------------
__global__ __launch_bounds__(256) void wconv_kernel(
    const float* __restrict__ W, _Float16* __restrict__ Wh) {
  const size_t i = ((size_t)blockIdx.x * 256 + threadIdx.x) * 8;
  f4 a0 = *(const f4*)(W + i);
  f4 a1 = *(const f4*)(W + i + 4);
  *(h8*)(Wh + i) =
      cat4(__builtin_convertvector(a0, h4), __builtin_convertvector(a1, h4));
}

// ---------------------------------------------------------------------------
// Projection GEMM: out = X @ W^T + bias, f16 head-split output.
//   VTRANS=0: out[((b*NH+h)*SEQ+s)*DHD + d]   (Q, K)
//   VTRANS=1: out[((b*NH+h)*DHD+d)*SEQ + s]   (V transposed for P*V B-frags)
// 256 thr = 8 waves (2M x 4N), block tile 64(M) x 128(N); each wave 32x32
// (4 wmma / k-step). A tile (64x32 f16) double-buffered in LDS.
// ---------------------------------------------------------------------------
template <int VTRANS>
__global__ __launch_bounds__(256) void proj_kernel(
    const float* __restrict__ X, const _Float16* __restrict__ Wh,
    const float* __restrict__ bias, _Float16* __restrict__ out) {
  constexpr int LDA = 40;  // padded halfs per LDS row (16B-aligned stride)
  __shared__ _Float16 ldsA[2][64 * LDA];

  const int tid  = threadIdx.x;
  const int lane = tid & 31, wave = tid >> 5;
  const int wm = wave >> 2, wn = wave & 3;          // 2(M) x 4(N) wave grid
  const int m0 = blockIdx.x * 64, n0 = blockIdx.y * 128;
  const int lrow = lane & 15, lg = lane >> 4;
  const int arow = tid >> 2, acol = (tid & 3) * 8;  // cooperative A stage

  const int n_base = n0 + wn * 32 + lrow;
  const _Float16* wb0 = Wh + (size_t)n_base * HID;
  const _Float16* wb1 = Wh + (size_t)(n_base + 16) * HID;
  const float* asrc = X + (size_t)(m0 + arow) * HID + acol;

  {  // prologue: stage k-block 0 into buffer 0
    f4 a0 = *(const f4*)asrc;
    f4 a1 = *(const f4*)(asrc + 4);
    *(h8*)&ldsA[0][arow * LDA + acol] =
        cat4(__builtin_convertvector(a0, h4), __builtin_convertvector(a1, h4));
  }

  v8f acc[2][2] = {{{}, {}}, {{}, {}}};
  for (int k0 = 0; k0 < HID; k0 += 32) {
    const int cur = (k0 >> 5) & 1;
    __syncthreads();  // buffer `cur` staged by everyone

    const bool more = (k0 + 32) < HID;
    f4 a0, a1;
    if (more) {  // prefetch next A tile into VGPRs (overlaps the wmma below)
      a0 = *(const f4*)(asrc + k0 + 32);
      a1 = *(const f4*)(asrc + k0 + 36);
    }

    const _Float16* ap0 = &ldsA[cur][(wm * 32 + lrow) * LDA];
    const _Float16* ap1 = &ldsA[cur][(wm * 32 + 16 + lrow) * LDA];
    h16 af0 = cat8(*(const h8*)(ap0 + lg * 16), *(const h8*)(ap0 + 8 + lg * 16));
    h16 af1 = cat8(*(const h8*)(ap1 + lg * 16), *(const h8*)(ap1 + 8 + lg * 16));
    h16 bf0 = load_b_f16(wb0 + k0 + lg * 16);
    h16 bf1 = load_b_f16(wb1 + k0 + lg * 16);
    acc[0][0] = wmma16(af0, bf0, acc[0][0]);
    acc[0][1] = wmma16(af0, bf1, acc[0][1]);
    acc[1][0] = wmma16(af1, bf0, acc[1][0]);
    acc[1][1] = wmma16(af1, bf1, acc[1][1]);

    if (more)
      *(h8*)&ldsA[1 - cur][arow * LDA + acol] = cat4(
          __builtin_convertvector(a0, h4), __builtin_convertvector(a1, h4));
  }

#pragma unroll
  for (int mi = 0; mi < 2; ++mi)
#pragma unroll
    for (int t = 0; t < 2; ++t) {
      const int n = n_base + t * 16;
      const float bval = bias[n];
      const int h = n >> 6, d = n & 63;
#pragma unroll
      for (int r = 0; r < 8; ++r) {
        const int m = m0 + wm * 32 + mi * 16 + r + 8 * lg;
        const int bb = m >> 11, s = m & (SEQ - 1);
        const _Float16 val = (_Float16)(acc[mi][t][r] + bval);
        if (VTRANS)
          out[((size_t)(bb * NH + h) * DHD + d) * SEQ + s] = val;
        else
          out[((size_t)(bb * NH + h) * SEQ + s) * DHD + d] = val;
      }
    }
}

// ---------------------------------------------------------------------------
// Flash attention: one 16-query tile per wave, 32 keys / iteration.
// Online softmax: running max via 16-lane butterfly; denominator from the
// matrix engine (accS = P x ones) -- every lane holds the row sum, no sum
// reduction/broadcast. Mask fused into per-row scale rs[r] in {0.125, 0}:
// masked row -> all-zero scores -> uniform softmax (== reference -1e9 row).
// hid is f16 [B,S,HID] row-major.
// ---------------------------------------------------------------------------
__global__ __launch_bounds__(128) void attn_kernel(
    const _Float16* __restrict__ Qh, const _Float16* __restrict__ Kh,
    const _Float16* __restrict__ Vt, const int* __restrict__ mask,
    _Float16* __restrict__ hid) {
  constexpr int LDP = 40;
  __shared__ _Float16 ldsP[4][16 * LDP];

  const int lane = threadIdx.x & 31, wave = threadIdx.x >> 5;
  const int lrow = lane & 15, lg = lane >> 4;
  const int tile = blockIdx.x * 4 + wave;
  const int qt = tile & (SEQ / 16 - 1);   // 128 query tiles per (b,h)
  const int bh = tile >> 7;
  const int b = bh >> 4, h = bh & 15;
  const int q0 = qt * 16;

  const _Float16* Qp = Qh + ((size_t)bh * SEQ + q0) * DHD;
  const _Float16* Kp = Kh + (size_t)bh * SEQ * DHD;
  const _Float16* Vp = Vt + (size_t)bh * DHD * SEQ;
  _Float16* ldp = ldsP[wave];

  h16 aQ[2];
#pragma unroll
  for (int c = 0; c < 2; ++c) {
    const _Float16* qp = Qp + (size_t)lrow * DHD + c * 32;
    aQ[c] = cat8(*(const h8*)(qp + lg * 16), *(const h8*)(qp + 8 + lg * 16));
  }

  float rs[8];  // per-row scale with mask fused: 0.125 (live) or 0 (masked)
  float mi[8];
#pragma unroll
  for (int r = 0; r < 8; ++r) {
    rs[r] = (mask[b * SEQ + q0 + r + 8 * lg] != 0) ? 0.125f : 0.0f;
    mi[r] = -INFINITY;
  }
  v8f accO[4];
#pragma unroll
  for (int nt = 0; nt < 4; ++nt) accO[nt] = {};
  v8f accS = {};  // row sums of P (softmax denominator), via ones-column wmma
  const h16 ones = splat_one16();

  for (int j0 = 0; j0 < SEQ; j0 += 32) {
    // ---- scores: two 16-key subtiles, K-dim = 64 -------------------------
    v8f sc[2];
#pragma unroll
    for (int kt = 0; kt < 2; ++kt) {
      v8f s = {};
#pragma unroll
      for (int c = 0; c < 2; ++c) {
        const _Float16* kp =
            Kp + (size_t)(j0 + kt * 16 + lrow) * DHD + c * 32 + lg * 16;
        s = wmma16(aQ[c], load_b_f16(kp), s);
      }
      sc[kt] = s;
    }

    // ---- online max + exp per query row ----------------------------------
    float p0[8], p1[8], fac[8];
#pragma unroll
    for (int r = 0; r < 8; ++r) {
      const float s0 = sc[0][r] * rs[r], s1 = sc[1][r] * rs[r];
      float tmax = fmaxf(s0, s1);
#pragma unroll
      for (int off = 1; off < 16; off <<= 1)
        tmax = fmaxf(tmax, __shfl_xor(tmax, off, 32));
      const float mnew = fmaxf(mi[r], tmax);
      fac[r] = __expf(mi[r] - mnew);
      mi[r] = mnew;
      p0[r] = __expf(s0 - mnew);
      p1[r] = __expf(s1 - mnew);
    }
#pragma unroll
    for (int nt = 0; nt < 4; ++nt)
#pragma unroll
      for (int r = 0; r < 8; ++r) accO[nt][r] *= fac[r];
#pragma unroll
    for (int r = 0; r < 8; ++r) accS[r] *= fac[r];

    // ---- P: C-layout -> LDS -> A-layout ----------------------------------
#pragma unroll
    for (int r = 0; r < 8; ++r) {
      const int row = r + 8 * lg;
      ldp[row * LDP + lrow]      = (_Float16)p0[r];
      ldp[row * LDP + 16 + lrow] = (_Float16)p1[r];
    }
    asm volatile("s_wait_dscnt 0x0" ::: "memory");
    h16 aP = cat8(*(const h8*)&ldp[lrow * LDP + lg * 16],
                  *(const h8*)&ldp[lrow * LDP + 8 + lg * 16]);

    // ---- O += P*V ; rowsum += P*1 ----------------------------------------
#pragma unroll
    for (int nt = 0; nt < 4; ++nt) {
      const _Float16* vp = Vp + (size_t)(nt * 16 + lrow) * SEQ + j0 + lg * 16;
      accO[nt] = wmma16(aP, load_b_f16(vp), accO[nt]);
    }
    accS = wmma16(aP, ones, accS);
  }

  // ---- normalize + store hidden [B,S,HID] f16 ----------------------------
#pragma unroll
  for (int r = 0; r < 8; ++r) {
    const float inv = 1.0f / accS[r];
    const size_t base =
        ((size_t)b * SEQ + q0 + r + 8 * lg) * HID + (size_t)h * DHD;
#pragma unroll
    for (int nt = 0; nt < 4; ++nt)
      hid[base + nt * 16 + lrow] = (_Float16)(accO[nt][r] * inv);
  }
}

// ---------------------------------------------------------------------------
// Output projection: d_out = hidden(f16) @ wo^T + bo, f32 output.
// A tile staged via CDNA5 async global->LDS DMA (ASYNCcnt), double-buffered:
// the async fill of the next buffer overlaps this buffer's wmma work and is
// only waited at the top of the next iteration.
// ---------------------------------------------------------------------------
__global__ __launch_bounds__(256) void outproj_kernel(
    const _Float16* __restrict__ Xh, const _Float16* __restrict__ Wh,
    const float* __restrict__ bias, float* __restrict__ out) {
  constexpr int LDA = 40;
  __shared__ _Float16 ldsA[2][64 * LDA];

  const int tid  = threadIdx.x;
  const int lane = tid & 31, wave = tid >> 5;
  const int wm = wave >> 2, wn = wave & 3;
  const int m0 = blockIdx.x * 64, n0 = blockIdx.y * 128;
  const int lrow = lane & 15, lg = lane >> 4;
  const int arow = tid >> 2, acol = (tid & 3) * 8;

  const int n_base = n0 + wn * 32 + lrow;
  const _Float16* wb0 = Wh + (size_t)n_base * HID;
  const _Float16* wb1 = Wh + (size_t)(n_base + 16) * HID;
  const _Float16* asrc = Xh + (size_t)(m0 + arow) * HID + acol;

  // low 32 bits of a generic LDS pointer = byte offset in the workgroup's
  // LDS allocation (aperture rules, ISA 10.2) -> async VDST operand.
  const unsigned ldsOff[2] = {
      (unsigned)(unsigned long long)(void*)&ldsA[0][arow * LDA + acol],
      (unsigned)(unsigned long long)(void*)&ldsA[1][arow * LDA + acol]};

  async_copy16(ldsOff[0], asrc);  // prologue fill of buffer 0

  v8f acc[2][2] = {{{}, {}}, {{}, {}}};
  for (int k0 = 0; k0 < HID; k0 += 32) {
    const int cur = (k0 >> 5) & 1;
    asm volatile("s_wait_asynccnt 0x0" ::: "memory");  // buffer `cur` landed
    __syncthreads();

    if (k0 + 32 < HID)  // kick off DMA for the next buffer; overlaps wmma
      async_copy16(ldsOff[1 - cur], asrc + k0 + 32);

    const _Float16* ap0 = &ldsA[cur][(wm * 32 + lrow) * LDA];
    const _Float16* ap1 = &ldsA[cur][(wm * 32 + 16 + lrow) * LDA];
    h16 af0 = cat8(*(const h8*)(ap0 + lg * 16), *(const h8*)(ap0 + 8 + lg * 16));
    h16 af1 = cat8(*(const h8*)(ap1 + lg * 16), *(const h8*)(ap1 + 8 + lg * 16));
    h16 bf0 = load_b_f16(wb0 + k0 + lg * 16);
    h16 bf1 = load_b_f16(wb1 + k0 + lg * 16);
    acc[0][0] = wmma16(af0, bf0, acc[0][0]);
    acc[0][1] = wmma16(af0, bf1, acc[0][1]);
    acc[1][0] = wmma16(af1, bf0, acc[1][0]);
    acc[1][1] = wmma16(af1, bf1, acc[1][1]);
  }

#pragma unroll
  for (int mi = 0; mi < 2; ++mi)
#pragma unroll
    for (int t = 0; t < 2; ++t) {
      const int n = n_base + t * 16;
      const float bval = bias[n];
#pragma unroll
      for (int r = 0; r < 8; ++r) {
        const int m = m0 + wm * 32 + mi * 16 + r + 8 * lg;
        out[(size_t)m * HID + n] = acc[mi][t][r] + bval;
      }
    }
}

// ---------------------------------------------------------------------------
extern "C" void kernel_launch(void* const* d_in, const int* in_sizes, int n_in,
                              void* d_out, int out_size, void* d_ws,
                              size_t ws_size, hipStream_t stream) {
  (void)in_sizes; (void)n_in; (void)out_size; (void)ws_size;
  const float* q    = (const float*)d_in[0];
  const float* k    = (const float*)d_in[1];
  const float* v    = (const float*)d_in[2];
  const int*   mask = (const int*)d_in[3];
  const float* wq   = (const float*)d_in[4];
  const float* bq   = (const float*)d_in[5];
  const float* wk   = (const float*)d_in[6];
  const float* bk   = (const float*)d_in[7];
  const float* wv   = (const float*)d_in[8];
  const float* bv   = (const float*)d_in[9];
  const float* wo   = (const float*)d_in[10];
  const float* bo   = (const float*)d_in[11];

  // ws layout (f16): Qh | Kh | Vt | hid | Whq | Whk | Whv | Whw  (72 MB)
  _Float16* Qh  = (_Float16*)d_ws;
  _Float16* Kh  = Qh + (size_t)MTOT * HID;
  _Float16* Vt  = Kh + (size_t)MTOT * HID;
  _Float16* hid = Vt + (size_t)MTOT * HID;
  _Float16* Whq = hid + (size_t)MTOT * HID;
  _Float16* Whk = Whq + (size_t)HID * HID;
  _Float16* Whv = Whk + (size_t)HID * HID;
  _Float16* Whw = Whv + (size_t)HID * HID;

  const int wcBlocks = (HID * HID) / (256 * 8);  // 512
  wconv_kernel<<<wcBlocks, 256, 0, stream>>>(wq, Whq);
  wconv_kernel<<<wcBlocks, 256, 0, stream>>>(wk, Whk);
  wconv_kernel<<<wcBlocks, 256, 0, stream>>>(wv, Whv);
  wconv_kernel<<<wcBlocks, 256, 0, stream>>>(wo, Whw);

  const dim3 gGemm(MTOT / 64, HID / 128);  // 128 x 8 blocks
  proj_kernel<0><<<gGemm, 256, 0, stream>>>(q, Whq, bq, Qh);
  proj_kernel<0><<<gGemm, 256, 0, stream>>>(k, Whk, bk, Kh);
  proj_kernel<1><<<gGemm, 256, 0, stream>>>(v, Whv, bv, Vt);

  const int nTiles = BATCH * NH * (SEQ / 16);  // 8192 query tiles
  attn_kernel<<<nTiles / 4, 128, 0, stream>>>(Qh, Kh, Vt, mask, hid);

  outproj_kernel<<<gGemm, 256, 0, stream>>>(hid, Whw, bo, (float*)d_out);
}
